// AttentionOnAttention_8083128451539
// MI455X (gfx1250) — matrix operand order repaired and checked by reading, not compiled, not run on hardware
//
#include <hip/hip_runtime.h>
#include <hip/hip_bf16.h>

// ---------------------------------------------------------------------------
// AttentionOnAttention for MI455X (gfx1250): all matmuls via v_wmma_f32_16x16x32_bf16
// Attention K/V tiles staged to LDS with GLOBAL_LOAD_ASYNC_TO_LDS_B128 (ASYNCcnt),
// double-buffered; fragments then read from LDS (ds_load_b128).
// B=2, N=2048, DIM=1024, H=16, DH=64, INNER=1024
// ---------------------------------------------------------------------------

#define BB    2
#define NN    2048
#define DIM   1024
#define HH    16
#define DHD   64
#define INNER 1024
#define ROWS  (BB * NN)           // 4096 token rows

typedef __attribute__((ext_vector_type(16))) __bf16 v16bf;
typedef __attribute__((ext_vector_type(8)))  float  v8f;

__device__ __forceinline__ unsigned short f32_to_bf16(float f) {
    unsigned int u = __float_as_uint(f);
    u += 0x7FFFu + ((u >> 16) & 1u);          // round-to-nearest-even
    return (unsigned short)(u >> 16);
}

__device__ __forceinline__ v8f wmma_bf16(v16bf a, v16bf b, v8f c) {
    // D(16x16,f32) = A(16x32,bf16) x B(32x16,bf16) + C
    return __builtin_amdgcn_wmma_f32_16x16x32_bf16(false, a, false, b,
                                                   (short)0, c, false, false);
}

// A fragment: source row-major (rows = M, contraction along columns), stride ld.
// lane l: row = l&15, halves 0..7 at K = 8*(l>>4), halves 8..15 at K = 16+8*(l>>4)
__device__ __forceinline__ v16bf load_frag_a(const unsigned short* base, int ld) {
    int lane = threadIdx.x & 31;
    const unsigned short* p = base + (lane & 15) * ld + (lane >> 4) * 8;
    union { v16bf v; uint4 q[2]; } f;
    f.q[0] = *(const uint4*)(p);
    f.q[1] = *(const uint4*)(p + 16);
    return f.v;
}

// B fragment: source stored N-major (rows = N, contraction along columns), stride ld.
// lane l: column n = l&15, halves 0..7 at K = 8*(l>>4), halves 8..15 at K = 16+8*(l>>4)
__device__ __forceinline__ v16bf load_frag_b(const unsigned short* base, int ld) {
    int lane = threadIdx.x & 31;
    const unsigned short* p = base + (lane & 15) * ld + (lane >> 4) * 8;
    union { v16bf v; uint4 q[2]; } f;
    f.q[0] = *(const uint4*)(p);
    f.q[1] = *(const uint4*)(p + 16);
    return f.v;
}

// ---- CDNA5 async global->LDS copy (ASYNCcnt path) -------------------------
// the vdst VGPR holds the LDS byte address; generic shared pointers carry the wave-relative
// LDS byte offset in their low 32 bits (aperture base low bits are zero).
__device__ __forceinline__ void async_copy_b128(unsigned lds_off, const void* gaddr) {
    asm volatile("global_load_async_to_lds_b128 %0, %1, off"
                 :: "v"(lds_off), "v"(gaddr) : "memory");
}
__device__ __forceinline__ void wait_async0() {
    asm volatile("s_wait_asynccnt 0x0" ::: "memory");
}
__device__ __forceinline__ unsigned lds_off(const void* shared_ptr) {
    return (unsigned)(unsigned long long)shared_ptr;
}

// ---------------------------------------------------------------------------
// fp32 -> bf16 straight copy (x, and the four 64x64 head weights)
// ---------------------------------------------------------------------------
__global__ void __launch_bounds__(256)
cvt_bf16_kernel(const float* __restrict__ src, unsigned short* __restrict__ dst, int n) {
    int i = blockIdx.x * 256 + threadIdx.x;
    if (i < n) dst[i] = f32_to_bf16(src[i]);
}

// fp32 -> bf16 with transpose: wt[n*K + k] = w[k*Nn + n]   (make weights N-major)
__global__ void __launch_bounds__(256)
cvt_bf16_t_kernel(const float* __restrict__ w, unsigned short* __restrict__ wt,
                  int K, int Nn) {
    int i = blockIdx.x * 256 + threadIdx.x;
    if (i < K * Nn) {
        int n = i / K, k = i - n * K;
        wt[i] = f32_to_bf16(w[(size_t)k * Nn + n]);
    }
}

// ---------------------------------------------------------------------------
// Fused QKV GEMM:  C(4096 x 3072) = Xb(4096x1024) @ [Wq | Wkv]  (WT is N-major)
//   cols [0,1024)    -> qb  token-major bf16
//   cols [1024,2048) -> kb  token-major bf16
//   cols [2048,3072) -> vT  feature-major bf16 (per batch: 1024 rows x 2048 tokens)
// Block = 256 thr = 8 waves; wave does 16 rows x 64 cols; grid (32, 48).
// ---------------------------------------------------------------------------
__global__ void __launch_bounds__(256)
gemm_qkv_kernel(const unsigned short* __restrict__ xb,
                const unsigned short* __restrict__ wt,
                unsigned short* __restrict__ qb,
                unsigned short* __restrict__ kb,
                unsigned short* __restrict__ vt) {
    const int wave = threadIdx.x >> 5;
    const int lane = threadIdx.x & 31;
    const int row0 = blockIdx.x * 128 + wave * 16;   // token-row tile
    const int n0   = blockIdx.y * 64;                // output column strip
    const int col  = lane & 15, hi = lane >> 4;

    v8f acc[4] = {};
    const unsigned short* arow = xb + (size_t)row0 * DIM;
    for (int k = 0; k < DIM; k += 32) {
        if (k + 32 < DIM) __builtin_prefetch(arow + k + 32, 0, 1);  // global_prefetch_b8
        v16bf a = load_frag_a(arow + k, DIM);
#pragma unroll
        for (int t = 0; t < 4; ++t) {
            v16bf b = load_frag_b(wt + (size_t)(n0 + t * 16) * DIM + k, DIM);
            acc[t] = wmma_bf16(a, b, acc[t]);
        }
    }

    const int region = n0 >> 10;       // 0 = q, 1 = k, 2 = v
    if (region < 2) {
        unsigned short* dst = (region == 0) ? qb : kb;
        const int nc0 = n0 & 1023;
#pragma unroll
        for (int t = 0; t < 4; ++t) {
            int nc = nc0 + t * 16 + col;
#pragma unroll
            for (int j = 0; j < 8; ++j)
                dst[(size_t)(row0 + hi * 8 + j) * INNER + nc] = f32_to_bf16(acc[t][j]);
        }
    } else {
        // v: transposed (feature-major) store, 8 consecutive tokens -> one b128
        const int bidx = row0 >> 11;               // batch
        const int tok0 = (row0 & (NN - 1)) + hi * 8;
#pragma unroll
        for (int t = 0; t < 4; ++t) {
            int f = (n0 - 2048) + t * 16 + col;    // feature 0..1023 (= h*64 + dh)
            unsigned int pk[4];
#pragma unroll
            for (int j = 0; j < 4; ++j) {
                unsigned int lo = f32_to_bf16(acc[t][2 * j]);
                unsigned int hw = f32_to_bf16(acc[t][2 * j + 1]);
                pk[j] = lo | (hw << 16);
            }
            *(uint4*)(vt + ((size_t)(bidx * INNER + f)) * NN + tok0) =
                make_uint4(pk[0], pk[1], pk[2], pk[3]);
        }
    }
}

// ---------------------------------------------------------------------------
// Flash attention (online softmax). One wave = 16 queries of one (b,h).
// S^T = K_tile x Q^T : D-layout of S^T == A-layout of P for the P@V WMMA
// when two 16-key tiles are packed into a 16x32 bf16 A fragment.
// K/V tiles (4KB each per 32-key step) are double-buffer staged to LDS with
// async global->LDS b128 copies shared by all 8 waves of the block.
// grid (16, 32): x = 128-query blocks, y = b*H + h.
// ---------------------------------------------------------------------------
__global__ void __launch_bounds__(256)
attn_kernel(const unsigned short* __restrict__ qb,
            const unsigned short* __restrict__ kb,
            const unsigned short* __restrict__ vt,
            unsigned short* __restrict__ ao) {
    __shared__ __attribute__((aligned(16))) unsigned short sk[2][32 * 64]; // keys x dh
    __shared__ __attribute__((aligned(16))) unsigned short sv[2][64 * 32]; // dh x keys

    const int tid  = threadIdx.x;
    const int wave = tid >> 5;
    const int lane = tid & 31;
    const int bh = blockIdx.y;
    const int b = bh >> 4, h = bh & 15;
    const int q0 = blockIdx.x * 128 + wave * 16;
    const int col = lane & 15, hi = lane >> 4;
    const float scale = 0.125f;   // DH^-0.5

    const unsigned short* qbase = qb + ((size_t)(b * NN + q0)) * INNER + h * DHD;
    const unsigned short* kbase = kb + ((size_t)(b * NN)) * INNER + h * DHD;
    const unsigned short* vbase = vt + ((size_t)(b * HH + h)) * DHD * NN;

    // cooperative staging slices: K tile 32 rows x 128B, V tile 64 rows x 64B
    const int krow = tid >> 3, kchunk = tid & 7;   // 8 elems (16B) per thread
    const int vrow = tid >> 2, vchunk = tid & 3;

    // Q^T as B-operand: N-major storage == token-major q. Covers dh 0..31 / 32..63.
    v16bf bq0 = load_frag_b(qbase, INNER);
    v16bf bq1 = load_frag_b(qbase + 32, INNER);

    v8f o[4] = {};
    float m_run = -1e30f, l_run = 0.0f;

    // prologue: stage tile 0 into buffer 0
    async_copy_b128(lds_off(&sk[0][krow * 64 + kchunk * 8]),
                    kbase + (size_t)krow * INNER + kchunk * 8);
    async_copy_b128(lds_off(&sv[0][vrow * 32 + vchunk * 8]),
                    vbase + (size_t)vrow * NN + vchunk * 8);

    for (int it = 0; it < NN / 32; ++it) {
        const int buf = it & 1;
        wait_async0();        // own slice of current tile complete
        __syncthreads();      // whole tile visible; everyone done reading prev buffer
        if (it + 1 < NN / 32) {
            const int key1 = (it + 1) * 32;
            async_copy_b128(lds_off(&sk[buf ^ 1][krow * 64 + kchunk * 8]),
                            kbase + (size_t)(key1 + krow) * INNER + kchunk * 8);
            async_copy_b128(lds_off(&sv[buf ^ 1][vrow * 32 + vchunk * 8]),
                            vbase + (size_t)vrow * NN + key1 + vchunk * 8);
        }
        const unsigned short* skc = sk[buf];
        const unsigned short* svc = sv[buf];

        // S^T for two 16-key tiles (A = K rows from LDS, contraction over dh)
        v8f s0 = {}, s1 = {};
        s0 = wmma_bf16(load_frag_a(skc, DHD), bq0, s0);
        s0 = wmma_bf16(load_frag_a(skc + 32, DHD), bq1, s0);
        s1 = wmma_bf16(load_frag_a(skc + 16 * DHD, DHD), bq0, s1);
        s1 = wmma_bf16(load_frag_a(skc + 16 * DHD + 32, DHD), bq1, s1);

        // per-query running max (lane = query; 16 key values per lane, 32 per lane-pair)
        float tmax = -1e30f;
#pragma unroll
        for (int j = 0; j < 8; ++j) {
            s0[j] *= scale; s1[j] *= scale;
            tmax = fmaxf(tmax, fmaxf(s0[j], s1[j]));
        }
        tmax = fmaxf(tmax, __shfl_xor(tmax, 16, 32));
        float m_new = fmaxf(m_run, tmax);
        float corr  = __expf(m_run - m_new);
        // rescale O: row m's factor lives at lane m (stats replicated in lane pair)
#pragma unroll
        for (int j = 0; j < 8; ++j) {
            float cf = __shfl(corr, hi * 8 + j, 32);
            o[0][j] *= cf; o[1][j] *= cf; o[2][j] *= cf; o[3][j] *= cf;
        }
        // probabilities -> bf16 A fragment (halves 0..7 = tile0, 8..15 = tile1)
        float psum = 0.0f;
        union { v16bf v; unsigned short hh[16]; } ap;
#pragma unroll
        for (int j = 0; j < 8; ++j) {
            float p0 = __expf(s0[j] - m_new);
            float p1 = __expf(s1[j] - m_new);
            psum += p0 + p1;
            ap.hh[j]     = f32_to_bf16(p0);
            ap.hh[8 + j] = f32_to_bf16(p1);
        }
        psum += __shfl_xor(psum, 16, 32);
        l_run = l_run * corr + psum;
        m_run = m_new;
        // O += P x V   (B from LDS V tile: rows = dh features, contiguous over keys)
#pragma unroll
        for (int t = 0; t < 4; ++t) {
            v16bf bv = load_frag_b(svc + (t * 16) * 32, 32);
            o[t] = wmma_bf16(ap.v, bv, o[t]);
        }
    }

    // normalize and store attn_out (token-major bf16)
    float inv = 1.0f / l_run;
    unsigned short* aobase = ao + ((size_t)(b * NN + q0)) * INNER + h * DHD;
#pragma unroll
    for (int j = 0; j < 8; ++j) {
        float sf = __shfl(inv, hi * 8 + j, 32);
        int m = hi * 8 + j;
#pragma unroll
        for (int t = 0; t < 4; ++t)
            aobase[(size_t)m * INNER + t * 16 + col] = f32_to_bf16(o[t][j] * sf);
    }
}

// ---------------------------------------------------------------------------
// Output/gate projection + sigmoid gating.
//   I = q@Wqo^T + ao@Wao^T + ob ;  G = q@Wqg^T + ao@Wag^T + gb ;  out = I*sigmoid(G)
// Row-major 64x64 weights ARE the N-major B layout for X@W^T. grid (16, 32).
// ---------------------------------------------------------------------------
__global__ void __launch_bounds__(256)
proj_kernel(const unsigned short* __restrict__ qb,
            const unsigned short* __restrict__ ao,
            const unsigned short* __restrict__ wqo,
            const unsigned short* __restrict__ wao,
            const unsigned short* __restrict__ wqg,
            const unsigned short* __restrict__ wag,
            const float* __restrict__ out_bias,
            const float* __restrict__ gate_bias,
            float* __restrict__ out) {
    const int wave = threadIdx.x >> 5;
    const int lane = threadIdx.x & 31;
    const int bh = blockIdx.y;
    const int b = bh >> 4, h = bh & 15;
    const int t0 = blockIdx.x * 128 + wave * 16;
    const int col = lane & 15, hi = lane >> 4;

    const unsigned short* qrow = qb + ((size_t)(b * NN + t0)) * INNER + h * DHD;
    const unsigned short* arow = ao + ((size_t)(b * NN + t0)) * INNER + h * DHD;
    v16bf aq0 = load_frag_a(qrow, INNER),     aq1 = load_frag_a(qrow + 32, INNER);
    v16bf aa0 = load_frag_a(arow, INNER),     aa1 = load_frag_a(arow + 32, INNER);

    v8f I[4] = {}, G[4] = {};
#pragma unroll
    for (int t = 0; t < 4; ++t) {
        I[t] = wmma_bf16(aq0, load_frag_b(wqo + t * 16 * DHD,      DHD), I[t]);
        I[t] = wmma_bf16(aq1, load_frag_b(wqo + t * 16 * DHD + 32, DHD), I[t]);
        I[t] = wmma_bf16(aa0, load_frag_b(wao + t * 16 * DHD,      DHD), I[t]);
        I[t] = wmma_bf16(aa1, load_frag_b(wao + t * 16 * DHD + 32, DHD), I[t]);
        G[t] = wmma_bf16(aq0, load_frag_b(wqg + t * 16 * DHD,      DHD), G[t]);
        G[t] = wmma_bf16(aq1, load_frag_b(wqg + t * 16 * DHD + 32, DHD), G[t]);
        G[t] = wmma_bf16(aa0, load_frag_b(wag + t * 16 * DHD,      DHD), G[t]);
        G[t] = wmma_bf16(aa1, load_frag_b(wag + t * 16 * DHD + 32, DHD), G[t]);
    }
#pragma unroll
    for (int t = 0; t < 4; ++t) {
        int d = t * 16 + col;
        float ob = out_bias[d], gb = gate_bias[d];
#pragma unroll
        for (int j = 0; j < 8; ++j) {
            float iv = I[t][j] + ob;
            float gv = G[t][j] + gb;
            float g  = 1.0f / (1.0f + __expf(-gv));
            int m = t0 + hi * 8 + j;
            out[((size_t)(b * NN + m)) * INNER + h * DHD + d] = iv * g;
        }
    }
}

// ---------------------------------------------------------------------------
extern "C" void kernel_launch(void* const* d_in, const int* in_sizes, int n_in,
                              void* d_out, int out_size, void* d_ws, size_t ws_size,
                              hipStream_t stream) {
    const float* x      = (const float*)d_in[0];
    const float* Wq     = (const float*)d_in[1];
    const float* Wkv    = (const float*)d_in[2];
    const float* Wq_out = (const float*)d_in[3];
    const float* Wa_out = (const float*)d_in[4];
    const float* o_bias = (const float*)d_in[5];
    const float* Wq_gat = (const float*)d_in[6];
    const float* Wa_gat = (const float*)d_in[7];
    const float* g_bias = (const float*)d_in[8];
    float* out = (float*)d_out;

    // workspace carve-out (~48 MB total; ws assumed large enough)
    char* ws = (char*)d_ws;
    size_t off = 0;
    auto alloc = [&](size_t bytes) -> void* {
        void* p = ws + off;
        off += (bytes + 255) & ~(size_t)255;
        return p;
    };
    unsigned short* xb  = (unsigned short*)alloc((size_t)ROWS * DIM * 2);      // 8 MB
    unsigned short* wt  = (unsigned short*)alloc((size_t)3072 * DIM * 2);      // 6 MB
    unsigned short* qb  = (unsigned short*)alloc((size_t)ROWS * INNER * 2);    // 8 MB
    unsigned short* kb  = (unsigned short*)alloc((size_t)ROWS * INNER * 2);    // 8 MB
    unsigned short* vT  = (unsigned short*)alloc((size_t)BB * INNER * NN * 2); // 8 MB
    unsigned short* ao  = (unsigned short*)alloc((size_t)ROWS * INNER * 2);    // 8 MB
    unsigned short* wqo = (unsigned short*)alloc(DHD * DHD * 2);
    unsigned short* wao = (unsigned short*)alloc(DHD * DHD * 2);
    unsigned short* wqg = (unsigned short*)alloc(DHD * DHD * 2);
    unsigned short* wag = (unsigned short*)alloc(DHD * DHD * 2);

    // 1) conversions
    {
        int n = ROWS * DIM;
        cvt_bf16_kernel<<<(n + 255) / 256, 256, 0, stream>>>(x, xb, n);
    }
    cvt_bf16_t_kernel<<<(DIM * INNER + 255) / 256, 256, 0, stream>>>(Wq, wt, DIM, INNER);
    cvt_bf16_t_kernel<<<(DIM * 2 * INNER + 255) / 256, 256, 0, stream>>>(
        Wkv, wt + (size_t)INNER * DIM, DIM, 2 * INNER);
    cvt_bf16_kernel<<<16, 256, 0, stream>>>(Wq_out, wqo, DHD * DHD);
    cvt_bf16_kernel<<<16, 256, 0, stream>>>(Wa_out, wao, DHD * DHD);
    cvt_bf16_kernel<<<16, 256, 0, stream>>>(Wq_gat, wqg, DHD * DHD);
    cvt_bf16_kernel<<<16, 256, 0, stream>>>(Wa_gat, wag, DHD * DHD);

    // 2) fused QKV GEMM: 4096 rows (32 blocks of 128) x 3072 cols (48 strips of 64)
    gemm_qkv_kernel<<<dim3(32, 48), 256, 0, stream>>>(xb, wt, qb, kb, vT);

    // 3) flash attention: 128 queries/block x 16 blocks, 32 (b,h) pairs
    attn_kernel<<<dim3(16, 32), 256, 0, stream>>>(qb, kb, vT, ao);

    // 4) output + gate projection with sigmoid gating
    proj_kernel<<<dim3(16, 32), 256, 0, stream>>>(qb, ao, wqo, wao, wqg, wag,
                                                  o_bias, g_bias, out);
}
